// ConvShare_39599598469533
// MI455X (gfx1250) — matrix operand order repaired and checked by reading, not compiled, not run on hardware
//
#include <hip/hip_runtime.h>

typedef __bf16 bf16_t;
typedef __attribute__((ext_vector_type(16))) __bf16 v16bf;
typedef __attribute__((ext_vector_type(8)))  __bf16 v8bf;
typedef __attribute__((ext_vector_type(8)))  float  v8f;

#define BATCH 8
#define SEQL  512
#define DIM   768
#define TAPS  12
#define ROWS  (BATCH * SEQL)          // 4096 (b,l) rows
#define KSL   (DIM / 32)              // 24 K-slices of 32 per 768-deep GEMM
#define PITCH 776                     // LDS row pitch in halves (768 + 8 pad)
#define MT    32                      // rows per workgroup (2 M-tiles)
#define XROWS (MT + TAPS - 1)         // 43 staged x rows

// ---------------- workspace layout (bf16 elements) ----------------
#define XB_OFF 0                                  // [ROWS][DIM]        bf16  6.29 MB
#define WC_OFF (ROWS * DIM)                       // [TAPS][DIM][DIM]   bf16 14.16 MB (k-major, o rows, d contiguous)
#define PW_OFF (WC_OFF + TAPS * DIM * DIM)        // [DIM][DIM]         bf16  1.18 MB (o rows, d contiguous)

// ---------------- one-time fp32 -> bf16 repack kernels ----------------
__global__ void cvt_x_kernel(const float* __restrict__ x, bf16_t* __restrict__ xb, int n) {
    int i = blockIdx.x * 256 + threadIdx.x;
    if (i < n) xb[i] = (__bf16)x[i];
}

// conv_w[o][d][k] fp32  ->  wc[k][o][d] bf16   (K=d contiguous for B-fragments)
__global__ void cvt_conv_kernel(const float* __restrict__ cw, bf16_t* __restrict__ wc, int n) {
    int i = blockIdx.x * 256 + threadIdx.x;
    if (i >= n) return;
    int k   = i / (DIM * DIM);
    int rem = i - k * (DIM * DIM);
    int o   = rem / DIM;
    int d   = rem - o * DIM;
    wc[i] = (__bf16)cw[(size_t)(o * DIM + d) * TAPS + k];
}

__global__ void cvt_pw_kernel(const float* __restrict__ pw, bf16_t* __restrict__ pwb, int n) {
    int i = blockIdx.x * 256 + threadIdx.x;
    if (i < n) pwb[i] = (__bf16)pw[i];
}

// Assemble a 16-half A fragment from two 16B chunks (layout: K 16h+0..7 | 16h+8..15 per lane half)
static __device__ inline v16bf frag2(const bf16_t* p0, const bf16_t* p1) {
    v8bf lo = *(const v8bf*)p0;
    v8bf hi = *(const v8bf*)p1;
    return __builtin_shufflevector(lo, hi, 0,1,2,3,4,5,6,7,8,9,10,11,12,13,14,15);
}

// ---------------- fused conv-cumsum-relu-projection kernel ----------------
// grid: 128 blocks (32 (b,l) rows each), block: 512 threads = 16 waves (wave32)
// wave w owns N columns [w*48, w*48+48) for BOTH 16-row M-halves:
// every B fragment is loaded once and feeds two WMMAs (halves L2 weight traffic).
__global__ __launch_bounds__(512)
void ConvShare_fused_kernel(const bf16_t* __restrict__ xb,
                            const bf16_t* __restrict__ wc,
                            const bf16_t* __restrict__ pw,
                            const float*  __restrict__ pb,
                            float* __restrict__ out) {
    __shared__ bf16_t xtile[XROWS * PITCH];  // rows l0 .. l0+42 of x (zero past L)   66.7 KB
    __shared__ bf16_t hbuf [MT * PITCH];     // relu(span_k), A-operand of projection 49.7 KB

    const int tid  = threadIdx.x;
    const int tile = blockIdx.x;                 // 32-row tile id
    const int b    = tile >> 4;                  // 512/32 = 16 tiles per batch
    const int l0   = (tile & 15) << 5;

    // ---- stage x rows into LDS (43 rows, zero past sequence end) ----
    for (int idx = tid; idx < XROWS * (DIM / 4); idx += 512) {
        int row  = idx / (DIM / 4);
        int col4 = (idx - row * (DIM / 4)) * 4;
        int l    = l0 + row;
        unsigned long long v = 0ULL;
        if (l < SEQL)
            v = *(const unsigned long long*)&xb[((size_t)(b * SEQL + l)) * DIM + col4];
        *(unsigned long long*)&xtile[row * PITCH + col4] = v;
    }
    __syncthreads();

    const int wave = tid >> 5;                   // 0..15
    const int lane = tid & 31;
    const int ln   = lane & 15;
    const int hi   = lane >> 4;                  // lane half selects K sub-range
    const int nb   = wave * 48;                  // this wave's 3 N-tiles

    // bias for this lane's output columns (C layout: N = lane%16 for both halves)
    float pbv[3];
#pragma unroll
    for (int t = 0; t < 3; ++t) pbv[t] = pb[nb + t * 16 + ln];

    // running span accumulators: [mh*3 + t], cumsum over taps lives here — never reset
    v8f c[6];
#pragma unroll
    for (int i = 0; i < 6; ++i) c[i] = (v8f)0.0f;

#pragma unroll 1
    for (int k = 0; k < TAPS; ++k) {
        // ---- stage 1: span += x[l+k : l+k+MT, :] @ conv_w[:,:,k]^T ----
        for (int kk = 0; kk < KSL; ++kk) {
            v16bf a[2];
#pragma unroll
            for (int mh = 0; mh < 2; ++mh) {
                const int abase = (mh * 16 + k + ln) * PITCH + kk * 32 + hi * 16;
                a[mh] = frag2(&xtile[abase], &xtile[abase + 8]);
            }
#pragma unroll
            for (int t = 0; t < 3; ++t) {
                const v16bf bm = *(const v16bf*)&wc[((size_t)(k * DIM + nb + t * 16 + ln)) * DIM
                                                    + kk * 32 + hi * 16];
#pragma unroll
                for (int mh = 0; mh < 2; ++mh)
                    c[mh * 3 + t] = __builtin_amdgcn_wmma_f32_16x16x32_bf16(
                        false, a[mh], false, bm, (short)0, c[mh * 3 + t], false, false);
            }
        }

        // ---- h = relu(span_k) -> LDS (bf16), becomes A of the projection ----
#pragma unroll
        for (int mh = 0; mh < 2; ++mh)
#pragma unroll
            for (int t = 0; t < 3; ++t) {
                const int col = nb + t * 16 + ln;
#pragma unroll
                for (int r = 0; r < 8; ++r)
                    hbuf[(mh * 16 + r + hi * 8) * PITCH + col] = (__bf16)fmaxf(c[mh * 3 + t][r], 0.0f);
            }
        __syncthreads();

        // ---- stage 2: out_k = relu(span_k) @ proj_w^T + bias ----
        v8f o2[6];
#pragma unroll
        for (int i = 0; i < 6; ++i) o2[i] = (v8f)0.0f;

        for (int kk = 0; kk < KSL; ++kk) {
            v16bf a[2];
#pragma unroll
            for (int mh = 0; mh < 2; ++mh) {
                const int abase = (mh * 16 + ln) * PITCH + kk * 32 + hi * 16;
                a[mh] = frag2(&hbuf[abase], &hbuf[abase + 8]);
            }
#pragma unroll
            for (int t = 0; t < 3; ++t) {
                const v16bf bm = *(const v16bf*)&pw[((size_t)(nb + t * 16 + ln)) * DIM
                                                    + kk * 32 + hi * 16];
#pragma unroll
                for (int mh = 0; mh < 2; ++mh)
                    o2[mh * 3 + t] = __builtin_amdgcn_wmma_f32_16x16x32_bf16(
                        false, a[mh], false, bm, (short)0, o2[mh * 3 + t], false, false);
            }
        }

        // ---- store out[b, l0+m, k, :] (two 64B runs per store instruction) ----
        const size_t row0 = (size_t)tile * MT;
#pragma unroll
        for (int mh = 0; mh < 2; ++mh)
#pragma unroll
            for (int t = 0; t < 3; ++t) {
#pragma unroll
                for (int r = 0; r < 8; ++r) {
                    const int m = mh * 16 + r + hi * 8;
                    out[((row0 + m) * TAPS + k) * DIM + nb + t * 16 + ln] = o2[mh * 3 + t][r] + pbv[t];
                }
            }
        __syncthreads();   // hbuf fully consumed before next tap overwrites it
    }
}

extern "C" void kernel_launch(void* const* d_in, const int* in_sizes, int n_in,
                              void* d_out, int out_size, void* d_ws, size_t ws_size,
                              hipStream_t stream) {
    const float* x      = (const float*)d_in[0];   // [8,512,768]
    const float* conv_w = (const float*)d_in[1];   // [768,768,12]
    const float* proj_w = (const float*)d_in[2];   // [768,768]
    const float* proj_b = (const float*)d_in[3];   // [768]
    float*       out    = (float*)d_out;           // [8,512,12,768]

    bf16_t* ws = (bf16_t*)d_ws;
    bf16_t* xb = ws + XB_OFF;
    bf16_t* wc = ws + WC_OFF;
    bf16_t* pw = ws + PW_OFF;

    const int nx = ROWS * DIM;             // 3,145,728
    const int nw = TAPS * DIM * DIM;       // 7,077,888
    const int np = DIM * DIM;              //   589,824

    cvt_x_kernel   <<<(nx + 255) / 256, 256, 0, stream>>>(x, xb, nx);
    cvt_conv_kernel<<<(nw + 255) / 256, 256, 0, stream>>>(conv_w, wc, nw);
    cvt_pw_kernel  <<<(np + 255) / 256, 256, 0, stream>>>(proj_w, pw, np);

    ConvShare_fused_kernel<<<ROWS / MT, 512, 0, stream>>>(xb, wc, pw, proj_b, out);
}
